// TransformerBlock_18743237280340
// MI455X (gfx1250) — compile-verified, matrix operand order checked by reading, not run on hardware
//
#include <hip/hip_runtime.h>
#include <hip/hip_bf16.h>
#include <math.h>
#include <stdint.h>

// ---------------------------------------------------------------------------
// Transformer block for MI455X (gfx1250).
//  - f16 WMMA GEMMs (v_wmma_f32_16x16x32_f16), f32 accumulate
//  - weights stored transposed (NxK) so B fragments are 128-bit LDS loads
//  - async global->LDS tile staging (global_load_async_to_lds_b128 +
//    s_wait_asynccnt) with double buffering
//  - fused QKV projection (N=768); flash attention with V^T layout, causal
//    mask only in the diagonal block, and WMMA-based row-sum accumulation
// D=256, S=2048, B=4, 8 heads x head_dim 32, hidden 1024.
// ---------------------------------------------------------------------------

typedef _Float16 h16 __attribute__((ext_vector_type(16)));
typedef _Float16 h8  __attribute__((ext_vector_type(8)));
typedef float    f8  __attribute__((ext_vector_type(8)));

#define D_MODEL 256
#define SEQ     2048
#define BATCH   4
#define NHEAD   8
#define HDIM    32
#define HID     1024
#define ROWS    (BATCH*SEQ)   // 8192
#define QKV_N   (3*D_MODEL)   // 768

// low 32 bits of a flat LDS pointer == LDS byte offset (aperture rules)
__device__ __forceinline__ uint32_t lds_addr32(const void* p) {
  return (uint32_t)(uintptr_t)p;
}

__device__ __forceinline__ void async_copy_b128(uint32_t lds_dst,
                                                const void* gsrc) {
  asm volatile("global_load_async_to_lds_b128 %0, %1, off"
               :: "v"(lds_dst), "v"(gsrc) : "memory");
}

__device__ __forceinline__ void wait_async0() {
  asm volatile("s_wait_asynccnt 0x0" ::: "memory");
}

// --------------------- weight convert + transpose ---------------------------
// src: [K][N] f32 row-major  ->  dst: [N][K] f16 row-major
__global__ void cvt_transpose_f16(const float* __restrict__ src,
                                  _Float16* __restrict__ dst, int K, int N) {
  int i = blockIdx.x * blockDim.x + threadIdx.x;
  if (i < K * N) {
    int k = i / N, n = i % N;
    dst[(size_t)n * K + k] = (_Float16)src[i];
  }
}

__global__ void pack_bias3(const float* __restrict__ a,
                           const float* __restrict__ b,
                           const float* __restrict__ c,
                           float* __restrict__ o) {
  int i = blockIdx.x * blockDim.x + threadIdx.x;
  if (i < D_MODEL) o[i] = a[i];
  else if (i < 2 * D_MODEL) o[i] = b[i - D_MODEL];
  else if (i < 3 * D_MODEL) o[i] = c[i - 2 * D_MODEL];
}

// V^T: qkv[B*S][768] (V at col 512+h*32+d) -> vt[((b*8+h)*32+d)*SEQ + s]
__global__ void build_vt(const _Float16* __restrict__ qkv,
                         _Float16* __restrict__ vt) {
  int i = blockIdx.x * blockDim.x + threadIdx.x;   // over ROWS*D_MODEL
  int s    = i & (SEQ - 1);
  int rest = i >> 11;
  int d    = rest & (HDIM - 1);
  int bh   = rest >> 5;
  int b    = bh >> 3, h = bh & 7;
  vt[i] = qkv[((size_t)(b * SEQ + s)) * QKV_N + 2 * D_MODEL + h * HDIM + d];
}

// ------------------------------ LayerNorm -----------------------------------
__global__ void layernorm_f16(const float* __restrict__ x,
                              const float* __restrict__ g,
                              const float* __restrict__ b,
                              _Float16* __restrict__ out) {
  const int row = blockIdx.x;
  const int tid = threadIdx.x;
  const float v = x[(size_t)row * D_MODEL + tid];
  float s1 = v, s2 = v * v;
#pragma unroll
  for (int off = 16; off >= 1; off >>= 1) {
    s1 += __shfl_xor(s1, off, 32);
    s2 += __shfl_xor(s2, off, 32);
  }
  __shared__ float red1[8], red2[8];
  const int wid = tid >> 5, lane = tid & 31;
  if (lane == 0) { red1[wid] = s1; red2[wid] = s2; }
  __syncthreads();
  float t1 = 0.f, t2 = 0.f;
#pragma unroll
  for (int i = 0; i < 8; i++) { t1 += red1[i]; t2 += red2[i]; }
  const float mu  = t1 * (1.0f / D_MODEL);
  const float var = t2 * (1.0f / D_MODEL) - mu * mu;
  const float r   = rsqrtf(var + 1e-5f);
  out[(size_t)row * D_MODEL + tid] = (_Float16)((v - mu) * r * g[tid] + b[tid]);
}

// ------------------------------ WMMA GEMM -----------------------------------
// C[M,N] = A[M,K](f16,rm) * WT[N,K](f16,rm)^T + bias.
//   EPI=0: ->f16   EPI=1: exact GELU ->f16   EPI=2: +resid ->f32
// Block tile 64x128, BK=32, 256 threads = 8 waves, each wave 32x32
// (4 v_wmma_f32_16x16x32_f16 per K-step). A and B^T tiles staged via async
// global->LDS b128 copies (ASYNCcnt), double buffered; all fragment reads
// from LDS are ds_load_b128.
template <int EPI>
__global__ void gemm_wmma(const _Float16* __restrict__ A,
                          const _Float16* __restrict__ WT,
                          const float* __restrict__ bias,
                          const float* __restrict__ resid,
                          void* __restrict__ outp, int M, int N, int K) {
  __shared__ _Float16 As[2][64 * 40];     // 64 m-rows  x 32 K (+8 pad)
  __shared__ _Float16 Bs[2][128 * 40];    // 128 n-rows x 32 K (+8 pad)

  const int tid  = threadIdx.x;
  const int wid  = tid >> 5, lane = tid & 31;
  const int l15  = lane & 15, hlf = lane >> 4;
  const int wm   = wid >> 2;              // 0..1 : 32-row slab
  const int wn   = wid & 3;               // 0..3 : 32-col slab
  const int m0   = blockIdx.y * 64, n0 = blockIdx.x * 128;

  // async-copy thread mapping (16B chunks, 8 halfs each)
  const int arow = tid >> 2, acol = (tid & 3) * 8;   // A:  64x32 = 256 chunks
  const int brow = tid >> 2, bcol = (tid & 3) * 8;   // BT:128x32 = 512 chunks

  auto issue_tile = [&](int buf, int k0) {
    async_copy_b128(lds_addr32(&As[buf][arow * 40 + acol]),
                    &A[(size_t)(m0 + arow) * K + k0 + acol]);
    async_copy_b128(lds_addr32(&Bs[buf][brow * 40 + bcol]),
                    &WT[(size_t)(n0 + brow) * K + k0 + bcol]);
    async_copy_b128(lds_addr32(&Bs[buf][(brow + 64) * 40 + bcol]),
                    &WT[(size_t)(n0 + brow + 64) * K + k0 + bcol]);
  };

  f8 acc[2][2] = {};
  int p = 0;
  issue_tile(0, 0);
  wait_async0();
  __syncthreads();

  for (int k0 = 0; k0 < K; k0 += 32) {
    if (k0 + 32 < K) issue_tile(p ^ 1, k0 + 32);   // prefetch next K tile

    // A fragments: lane l -> row (wm*32 + mi*16 + l&15),
    // K chunks {8h..8h+7} and {16+8h..23+8h}
    h16 af[2];
#pragma unroll
    for (int mi = 0; mi < 2; mi++) {
      const int m = wm * 32 + mi * 16 + l15;
      const h8 lo = *(const h8*)&As[p][m * 40 + hlf * 8];
      const h8 hi = *(const h8*)&As[p][m * 40 + 16 + hlf * 8];
#pragma unroll
      for (int j = 0; j < 8; j++) { af[mi][j] = lo[j]; af[mi][j + 8] = hi[j]; }
    }
    // B fragments: lane l -> col (wn*32 + ni*16 + l&15); contiguous K in LDS
    h16 bf[2];
#pragma unroll
    for (int ni = 0; ni < 2; ni++) {
      const int n = wn * 32 + ni * 16 + l15;
      const h8 lo = *(const h8*)&Bs[p][n * 40 + hlf * 8];
      const h8 hi = *(const h8*)&Bs[p][n * 40 + 16 + hlf * 8];
#pragma unroll
      for (int j = 0; j < 8; j++) { bf[ni][j] = lo[j]; bf[ni][j + 8] = hi[j]; }
    }
#pragma unroll
    for (int mi = 0; mi < 2; mi++)
#pragma unroll
      for (int ni = 0; ni < 2; ni++)
        acc[mi][ni] = __builtin_amdgcn_wmma_f32_16x16x32_f16(
            false, af[mi], false, bf[ni], (short)0, acc[mi][ni], false, false);

    wait_async0();      // next buffer landed
    __syncthreads();    // everyone done reading buffer p
    p ^= 1;
  }

  // Epilogue. C layout: (r, lane) -> row r + 8*(lane>>4), col lane&15.
#pragma unroll
  for (int mi = 0; mi < 2; mi++) {
#pragma unroll
    for (int ni = 0; ni < 2; ni++) {
#pragma unroll
      for (int r = 0; r < 8; r++) {
        const int grow = m0 + wm * 32 + mi * 16 + r + 8 * hlf;
        const int gcol = n0 + wn * 32 + ni * 16 + l15;
        float v0 = acc[mi][ni][r] + bias[gcol];
        const size_t i0 = (size_t)grow * N + gcol;
        if (EPI == 0) {
          ((_Float16*)outp)[i0] = (_Float16)v0;
        } else if (EPI == 1) {
          v0 = 0.5f * v0 * (1.0f + erff(v0 * 0.70710678118f));
          ((_Float16*)outp)[i0] = (_Float16)v0;
        } else {
          ((float*)outp)[i0] = v0 + resid[i0];
        }
      }
    }
  }
}

// ---------------------------- Flash attention -------------------------------
// Per-wave softmax state for one 16-row q tile. The softmax denominator is
// accumulated with a WMMA against an all-ones B matrix (accl = P*1 + accl),
// so no lane reductions are needed for the row sums.
struct AttnState {
  h16 qa;
  float mrow[8];
  f8  acc0, acc1;   // output accumulators (head dims 0-15, 16-31)
  f8  accl;         // row-sum accumulator (every lane of a row holds the sum)
};

// Process one 32-key block. MASKED=true only for the diagonal block.
template <bool MASKED>
__device__ __forceinline__ void attn_block(AttnState& st,
                                           const _Float16* __restrict__ kp,
                                           const _Float16* __restrict__ vtp,
                                           _Float16* __restrict__ Pw,
                                           int kb, int q0, int l15, int hlf) {
  const f8 zero = {};
  const float scale = 0.0625f;   // 1/sqrt(256) per reference

  // ---- scores: two 16x16 tiles over keys [kb, kb+32) ----
  f8 s[2];
#pragma unroll
  for (int t = 0; t < 2; t++) {
    const int key = kb + t * 16 + l15;   // B operand: lane -> key column
    const h8 lo = *(const h8*)&kp[(size_t)key * QKV_N + hlf * 8];
    const h8 hi = *(const h8*)&kp[(size_t)key * QKV_N + 16 + hlf * 8];
    h16 kf;
#pragma unroll
    for (int j = 0; j < 8; j++) { kf[j] = lo[j]; kf[j + 8] = hi[j]; }
    s[t] = __builtin_amdgcn_wmma_f32_16x16x32_f16(false, st.qa, false, kf,
                                                  (short)0, zero, false, false);
  }
  // ---- online softmax per row (row lives in a 16-lane half) ----
#pragma unroll
  for (int r = 0; r < 8; r++) {
    float s0 = s[0][r] * scale;
    float s1 = s[1][r] * scale;
    if (MASKED) {
      const int row = q0 + r + 8 * hlf;
      if (kb + l15 > row)      s0 = -1.0e30f;
      if (kb + 16 + l15 > row) s1 = -1.0e30f;
    }
    float mx = fmaxf(s0, s1);
    mx = fmaxf(mx, __shfl_xor(mx, 1, 32));
    mx = fmaxf(mx, __shfl_xor(mx, 2, 32));
    mx = fmaxf(mx, __shfl_xor(mx, 4, 32));
    mx = fmaxf(mx, __shfl_xor(mx, 8, 32));
    const float mnew = fmaxf(st.mrow[r], mx);
    const float efac = __expf(st.mrow[r] - mnew);
    st.mrow[r] = mnew;
    const float p0 = __expf(s0 - mnew);
    const float p1 = __expf(s1 - mnew);
    st.acc0[r] *= efac;
    st.acc1[r] *= efac;
    st.accl[r] *= efac;
    const int R = r + 8 * hlf;          // C-layout row of this element
    Pw[R * 40 + l15]      = (_Float16)p0;
    Pw[R * 40 + 16 + l15] = (_Float16)p1;
  }
  // ---- reload P in A-operand layout (same wave; LDS ops are in-order) ----
  h16 pa;
  {
    const h8 lo = *(const h8*)&Pw[l15 * 40 + hlf * 8];
    const h8 hi = *(const h8*)&Pw[l15 * 40 + 16 + hlf * 8];
#pragma unroll
    for (int j = 0; j < 8; j++) { pa[j] = lo[j]; pa[j + 8] = hi[j]; }
  }
  // ---- V fragments from V^T: per-lane head-dim row, contiguous keys ----
  h16 vf[2];
#pragma unroll
  for (int ni = 0; ni < 2; ni++) {
    const int hd = ni * 16 + l15;
    const h8 lo = *(const h8*)&vtp[(size_t)hd * SEQ + kb + hlf * 8];
    const h8 hi = *(const h8*)&vtp[(size_t)hd * SEQ + kb + 16 + hlf * 8];
#pragma unroll
    for (int j = 0; j < 8; j++) { vf[ni][j] = lo[j]; vf[ni][j + 8] = hi[j]; }
  }
  st.acc0 = __builtin_amdgcn_wmma_f32_16x16x32_f16(false, pa, false, vf[0],
                                                   (short)0, st.acc0, false, false);
  st.acc1 = __builtin_amdgcn_wmma_f32_16x16x32_f16(false, pa, false, vf[1],
                                                   (short)0, st.acc1, false, false);
  // ---- row sums via WMMA against all-ones B (no lane reductions) ----
  h16 ones;
#pragma unroll
  for (int j = 0; j < 16; j++) ones[j] = (_Float16)1.0f;
  st.accl = __builtin_amdgcn_wmma_f32_16x16x32_f16(false, pa, false, ones,
                                                   (short)0, st.accl, false, false);
}

// Q/K read from packed qkv [ROWS][768]; V read from V^T [B,H,HDIM,SEQ] so
// every fragment is two 128-bit loads. One wave per 16-row q tile.
// Unmasked steady-state blocks + one masked diagonal block.
__global__ void attn_wmma(const _Float16* __restrict__ qkv,
                          const _Float16* __restrict__ vt,
                          _Float16* __restrict__ o) {
  __shared__ _Float16 P[4][16 * 40];  // per-wave P staging (16x32 + pad)
  const int tid = threadIdx.x;
  const int wid = tid >> 5, lane = tid & 31;
  const int l15 = lane & 15, hlf = lane >> 4;
  const int bh  = blockIdx.x;
  const int b   = bh >> 3, h = bh & 7;
  const int q0  = (blockIdx.y * 4 + wid) * 16;
  const size_t base3 = ((size_t)b * SEQ) * QKV_N + h * HDIM;
  const _Float16* qp  = qkv + base3;                       // stride 768
  const _Float16* kp  = qkv + base3 + D_MODEL;             // stride 768
  const _Float16* vtp = vt + ((size_t)bh * HDIM) * SEQ;    // [HDIM][SEQ]
  _Float16*       op  = o + ((size_t)b * SEQ) * D_MODEL + h * HDIM;
  _Float16*       Pw  = &P[wid][0];

  AttnState st;
  {
    const int qrow = q0 + l15;
    const h8 lo = *(const h8*)&qp[(size_t)qrow * QKV_N + hlf * 8];
    const h8 hi = *(const h8*)&qp[(size_t)qrow * QKV_N + 16 + hlf * 8];
#pragma unroll
    for (int j = 0; j < 8; j++) { st.qa[j] = lo[j]; st.qa[j + 8] = hi[j]; }
  }
#pragma unroll
  for (int r = 0; r < 8; r++) st.mrow[r] = -3.0e38f;
  st.acc0 = f8{}; st.acc1 = f8{}; st.accl = f8{};

  // steady state: blocks fully below the diagonal (no masking needed)
  int kb = 0;
  for (; kb + 32 <= q0; kb += 32) {
    __builtin_prefetch(&kp[(size_t)(kb + 32 + l15) * QKV_N], 0, 1);
    __builtin_prefetch(&vtp[(size_t)lane * SEQ + kb + 32], 0, 1);
    attn_block<false>(st, kp, vtp, Pw, kb, q0, l15, hlf);
  }
  // diagonal block (keys [kb, kb+32) overlap rows [q0, q0+16))
  attn_block<true>(st, kp, vtp, Pw, kb, q0, l15, hlf);

#pragma unroll
  for (int r = 0; r < 8; r++) {
    const int row = q0 + r + 8 * hlf;
    const float inv = 1.0f / st.accl[r];
    op[(size_t)row * D_MODEL + l15]      = (_Float16)(st.acc0[r] * inv);
    op[(size_t)row * D_MODEL + 16 + l15] = (_Float16)(st.acc1[r] * inv);
  }
}

// ------------------------------ launcher ------------------------------------
extern "C" void kernel_launch(void* const* d_in, const int* in_sizes, int n_in,
                              void* d_out, int out_size, void* d_ws,
                              size_t ws_size, hipStream_t stream) {
  const float* x      = (const float*)d_in[0];
  const float* ln1_g  = (const float*)d_in[1];
  const float* ln1_b  = (const float*)d_in[2];
  const float* wq     = (const float*)d_in[3];
  const float* bq     = (const float*)d_in[4];
  const float* wk     = (const float*)d_in[5];
  const float* bk     = (const float*)d_in[6];
  const float* wv     = (const float*)d_in[7];
  const float* bv     = (const float*)d_in[8];
  const float* wo     = (const float*)d_in[9];
  const float* bo     = (const float*)d_in[10];
  const float* ln2_g  = (const float*)d_in[11];
  const float* ln2_b  = (const float*)d_in[12];
  const float* w_fc   = (const float*)d_in[13];
  const float* b_fc   = (const float*)d_in[14];
  const float* w_proj = (const float*)d_in[15];
  const float* b_proj = (const float*)d_in[16];
  float* out = (float*)d_out;

  // Workspace carve (256B aligned)
  char* ws = (char*)d_ws;
  auto carve = [&](size_t bytes) -> void* {
    void* p = ws;
    ws += (bytes + 255) & ~(size_t)255;
    return p;
  };
  _Float16* wtqkv_h = (_Float16*)carve((size_t)QKV_N * D_MODEL * 2);
  _Float16* wt_o    = (_Float16*)carve((size_t)D_MODEL * D_MODEL * 2);
  _Float16* wt_fc   = (_Float16*)carve((size_t)HID * D_MODEL * 2);
  _Float16* wt_pr   = (_Float16*)carve((size_t)D_MODEL * HID * 2);
  float*    bqkv    = (float*)   carve((size_t)QKV_N * 4);
  _Float16* ln1_h   = (_Float16*)carve((size_t)ROWS * D_MODEL * 2);
  _Float16* qkv_h   = (_Float16*)carve((size_t)ROWS * QKV_N * 2);
  _Float16* vt_h    = (_Float16*)carve((size_t)ROWS * D_MODEL * 2);
  _Float16* attn_h  = (_Float16*)carve((size_t)ROWS * D_MODEL * 2);
  _Float16* ln2_h   = (_Float16*)carve((size_t)ROWS * D_MODEL * 2);
  _Float16* fc_h    = (_Float16*)carve((size_t)ROWS * HID * 2);
  float*    x1_f    = (float*)   carve((size_t)ROWS * D_MODEL * 4);

  // 1) weights -> f16, transposed to [N][K]; pack QKV bias
  const int nw  = D_MODEL * D_MODEL;          // 65536
  const int nfc = D_MODEL * HID;              // 262144
  cvt_transpose_f16<<<(nw + 255) / 256, 256, 0, stream>>>(
      wq, wtqkv_h + 0 * (size_t)D_MODEL * D_MODEL, D_MODEL, D_MODEL);
  cvt_transpose_f16<<<(nw + 255) / 256, 256, 0, stream>>>(
      wk, wtqkv_h + 1 * (size_t)D_MODEL * D_MODEL, D_MODEL, D_MODEL);
  cvt_transpose_f16<<<(nw + 255) / 256, 256, 0, stream>>>(
      wv, wtqkv_h + 2 * (size_t)D_MODEL * D_MODEL, D_MODEL, D_MODEL);
  cvt_transpose_f16<<<(nw + 255) / 256, 256, 0, stream>>>(
      wo, wt_o, D_MODEL, D_MODEL);
  cvt_transpose_f16<<<(nfc + 255) / 256, 256, 0, stream>>>(
      w_fc, wt_fc, D_MODEL, HID);      // [256][1024] -> [1024][256]
  cvt_transpose_f16<<<(nfc + 255) / 256, 256, 0, stream>>>(
      w_proj, wt_pr, HID, D_MODEL);    // [1024][256] -> [256][1024]
  pack_bias3<<<3, 256, 0, stream>>>(bq, bk, bv, bqkv);

  // 2) LN1
  layernorm_f16<<<ROWS, 256, 0, stream>>>(x, ln1_g, ln1_b, ln1_h);

  // 3) fused QKV projection (M=8192, N=768, K=256)
  dim3 gqkv(QKV_N / 128, ROWS / 64);          // (6, 128)
  gemm_wmma<0><<<gqkv, 256, 0, stream>>>(ln1_h, wtqkv_h, bqkv, nullptr, qkv_h,
                                         ROWS, QKV_N, D_MODEL);

  // 4) V^T for attention, then causal flash attention
  build_vt<<<(ROWS * D_MODEL) / 256, 256, 0, stream>>>(qkv_h, vt_h);
  dim3 ga(BATCH * NHEAD, SEQ / 64);           // (32, 32)
  attn_wmma<<<ga, 128, 0, stream>>>(qkv_h, vt_h, attn_h);

  // 5) O projection + residual -> x1 (f32)
  dim3 gq(D_MODEL / 128, ROWS / 64);          // (2, 128)
  gemm_wmma<2><<<gq, 256, 0, stream>>>(attn_h, wt_o, bo, x, x1_f,
                                       ROWS, D_MODEL, D_MODEL);

  // 6) LN2
  layernorm_f16<<<ROWS, 256, 0, stream>>>(x1_f, ln2_g, ln2_b, ln2_h);

  // 7) FC + exact GELU (M=8192, N=1024, K=256)
  dim3 gfc(HID / 128, ROWS / 64);             // (8, 128)
  gemm_wmma<1><<<gfc, 256, 0, stream>>>(ln2_h, wt_fc, b_fc, nullptr, fc_h,
                                        ROWS, HID, D_MODEL);

  // 8) down-proj + residual -> d_out (M=8192, N=256, K=1024)
  gemm_wmma<2><<<gq, 256, 0, stream>>>(fc_h, wt_pr, b_proj, x1_f, out,
                                       ROWS, D_MODEL, HID);
}
